// VectorQuantizerMultiHead_72026601554053
// MI455X (gfx1250) — matrix operand order repaired and checked by reading, not compiled, not run on hardware
//
#include <hip/hip_runtime.h>
#include <hip/hip_bf16.h>

// ---------------------------------------------------------------------------
// VectorQuantizerMultiHead for MI455X (gfx1250, wave32, WMMA)
//   inputs   : [8,2048,1024] f32  ->  N=16384 rows, H=4 heads, dh=256
//   codebook : [4,1024,256] f32   ->  K=1024 codes per head
//   out      : [1 + 16777216] f32 : loss scalar, then quantized (straight-through)
// Distance argmin uses bf16x2-split WMMA (hi*hi + hi*lo + lo*hi) for ~f32
// accuracy on the dot products at bf16 matrix-pipe throughput.
// ---------------------------------------------------------------------------

#define NROWS  16384
#define DH     256
#define KCODES 1024
#define HEADS  4
#define NELEM  16777216          // 8*2048*1024
#define GATHER_BLOCKS 65536      // NELEM / 256

typedef __bf16 bf16_t;
typedef bf16_t v16bf __attribute__((ext_vector_type(16)));
typedef float  v8f   __attribute__((ext_vector_type(8)));

// ---------------------------------------------------------------------------
// Kernel 0: codebook f32 -> bf16 hi/lo split + per-code squared norm.
// One wave per code row; lane handles 8 strided d positions (coalesced).
// ---------------------------------------------------------------------------
__global__ __launch_bounds__(256) void vq_prep_kernel(
    const float* __restrict__ codebook,
    bf16_t* __restrict__ cb_hi,
    bf16_t* __restrict__ cb_lo,
    float* __restrict__ c_sq)
{
    const int lane = threadIdx.x & 31;
    const int wave = threadIdx.x >> 5;
    const int row  = blockIdx.x * 8 + wave;      // [0, HEADS*KCODES)
    const float* src = codebook + (size_t)row * DH;
    bf16_t* dh = cb_hi + (size_t)row * DH;
    bf16_t* dl = cb_lo + (size_t)row * DH;

    float s = 0.0f;
#pragma unroll
    for (int j = 0; j < 8; ++j) {
        const int d = j * 32 + lane;
        const float f = src[d];
        s += f * f;
        const bf16_t hi = (bf16_t)f;
        dh[d] = hi;
        dl[d] = (bf16_t)(f - (float)hi);
    }
#pragma unroll
    for (int mask = 16; mask >= 1; mask >>= 1)
        s += __shfl_xor(s, mask, 32);
    if (lane == 0) c_sq[row] = s;
}

// ---------------------------------------------------------------------------
// Kernel 1: per-(head,row) nearest-code search via WMMA.
// Block = 256 threads = 8 waves; each wave owns 16 rows of one head and sweeps
// all 1024 codes in 16-code tiles. dot accumulated over dh=256 as 8 chunks of
// K=32 with 3 bf16 WMMAs each (bf16x2 split). A fragments resident in VGPRs
// (__launch_bounds__(256) gives the compiler the register budget: ~200 VGPRs
// per wave, no scratch spills).
//
// VGPR layouts used here (16-bit A 16x32 / B 32x16, wave32; contiguous k, not the ISA table order):
//   A: lane L holds row m=L&15; K-halves kk+0..15, kk = (L<16)?0:16
//      -> one contiguous run of the input row.
//   B: lane L holds col n=L&15; K-halves kb+0..15, kb = (L<16)?0:16
//      -> one contiguous 32B run of codebook[n] (B = codebook^T).
//   C/D: lane L col n=L&15; VGPR r -> row M = r + (L<16 ? 0 : 8).
// ---------------------------------------------------------------------------
__global__ __launch_bounds__(256) void vq_argmin_kernel(
    const float* __restrict__ inputs,
    const bf16_t* __restrict__ cb_hi,
    const bf16_t* __restrict__ cb_lo,
    const float* __restrict__ c_sq,
    int* __restrict__ idx_out)
{
    const int lane   = threadIdx.x & 31;
    const int wave   = threadIdx.x >> 5;
    const int h      = blockIdx.y;
    const int n0     = blockIdx.x * 128 + wave * 16;   // first of 16 rows
    const int m      = lane & 15;
    const bool hiH   = lane >= 16;
    const int kk     = hiH ? 16 : 0;                   // A K-offset
    const int kb     = hiH ? 16 : 0;                   // B K-offset

    // ---- load my input row and build resident bf16 hi/lo A fragments ----
    const float* xrow = inputs + (size_t)(n0 + m) * (HEADS * DH) + h * DH;
    v16bf a_hi[8], a_lo[8];
#pragma unroll
    for (int c = 0; c < 8; ++c) {
        const float* p0 = xrow + 32 * c + kk;          // K kk+0..7
        const float* p1 = xrow + 32 * c + kk + 8;      // K kk+8..15
#pragma unroll
        for (int j = 0; j < 8; ++j) {
            const float f = p0[j];
            const bf16_t fh = (bf16_t)f;
            a_hi[c][j] = fh;
            a_lo[c][j] = (bf16_t)(f - (float)fh);
            const float g = p1[j];
            const bf16_t gh = (bf16_t)g;
            a_hi[c][8 + j] = gh;
            a_lo[c][8 + j] = (bf16_t)(g - (float)gh);
        }
    }

    float bestVal[8];
    int   bestIdx[8];
#pragma unroll
    for (int r = 0; r < 8; ++r) { bestVal[r] = __builtin_inff(); bestIdx[r] = 0; }

    const size_t headOff = (size_t)h * KCODES * DH;

    for (int t = 0; t < KCODES / 16; ++t) {
        const int code = t * 16 + m;                   // my B column's code id
        const bf16_t* bh = cb_hi + headOff + (size_t)code * DH + kb;
        const bf16_t* bl = cb_lo + headOff + (size_t)code * DH + kb;

        // keep the B stream ahead of the matrix pipe (global_prefetch_b8)
        if (t + 1 < KCODES / 16) {
            __builtin_prefetch(bh + 16 * DH, 0, 0);
            __builtin_prefetch(bl + 16 * DH, 0, 0);
        }

        v8f acc = {0.f, 0.f, 0.f, 0.f, 0.f, 0.f, 0.f, 0.f};
#pragma unroll
        for (int c = 0; c < 8; ++c) {
            const v16bf Bh = *(const v16bf*)(bh + 32 * c);
            const v16bf Bl = *(const v16bf*)(bl + 32 * c);
            acc = __builtin_amdgcn_wmma_f32_16x16x32_bf16(
                      false, a_hi[c], false, Bh, (short)0, acc, false, false);
            acc = __builtin_amdgcn_wmma_f32_16x16x32_bf16(
                      false, a_hi[c], false, Bl, (short)0, acc, false, false);
            acc = __builtin_amdgcn_wmma_f32_16x16x32_bf16(
                      false, a_lo[c], false, Bh, (short)0, acc, false, false);
        }

        const float csq = c_sq[h * KCODES + code];
#pragma unroll
        for (int r = 0; r < 8; ++r) {
            const float d = csq - 2.0f * acc[r];       // + x_sq (const per row)
            if (d < bestVal[r]) { bestVal[r] = d; bestIdx[r] = code; }
        }
    }

    // ---- argmin reduce across the 16 lanes holding each row ----
#pragma unroll
    for (int mask = 1; mask < 16; mask <<= 1) {
#pragma unroll
        for (int r = 0; r < 8; ++r) {
            const float ov = __shfl_xor(bestVal[r], mask, 32);
            const int   oi = __shfl_xor(bestIdx[r], mask, 32);
            if (ov < bestVal[r] || (ov == bestVal[r] && oi < bestIdx[r])) {
                bestVal[r] = ov; bestIdx[r] = oi;
            }
        }
    }
    if (m == 0) {
        const int rowBase = n0 + (hiH ? 8 : 0);
#pragma unroll
        for (int r = 0; r < 8; ++r)
            idx_out[h * NROWS + rowBase + r] = bestIdx[r];
    }
}

// ---------------------------------------------------------------------------
// Kernel 2: gather quantized output (f32 codebook) + per-block SSE partials.
// ---------------------------------------------------------------------------
__global__ __launch_bounds__(256) void vq_gather_kernel(
    const float* __restrict__ inputs,
    const float* __restrict__ codebook,
    const int* __restrict__ idx,
    float* __restrict__ q_out,
    float* __restrict__ partials)
{
    __shared__ float smem[256];
    const size_t e  = (size_t)blockIdx.x * 256 + threadIdx.x;
    const int n     = (int)(e >> 10);
    const int col   = (int)(e & 1023);
    const int h     = col >> 8;
    const int d     = col & 255;
    const int k     = idx[h * NROWS + n];
    const float q   = codebook[((size_t)(h * KCODES + k)) * DH + d];
    const float x   = inputs[e];
    q_out[e] = q;                                      // straight-through value
    const float diff = q - x;
    smem[threadIdx.x] = diff * diff;
    __syncthreads();
    for (int s = 128; s > 0; s >>= 1) {
        if (threadIdx.x < (unsigned)s) smem[threadIdx.x] += smem[threadIdx.x + s];
        __syncthreads();
    }
    if (threadIdx.x == 0) partials[blockIdx.x] = smem[0];
}

// ---------------------------------------------------------------------------
// Kernel 3: deterministic fixed-order reduction of partials -> loss scalar.
// ---------------------------------------------------------------------------
__global__ __launch_bounds__(256) void vq_loss_kernel(
    const float* __restrict__ partials,
    float* __restrict__ loss)
{
    __shared__ float smem[256];
    float s = 0.0f;
    for (int i = threadIdx.x; i < GATHER_BLOCKS; i += 256) s += partials[i];
    smem[threadIdx.x] = s;
    __syncthreads();
    for (int st = 128; st > 0; st >>= 1) {
        if (threadIdx.x < (unsigned)st) smem[threadIdx.x] += smem[threadIdx.x + st];
        __syncthreads();
    }
    if (threadIdx.x == 0)
        loss[0] = 0.25f * smem[0] / (float)NELEM;      // COMMITMENT_COST * mean
}

// ---------------------------------------------------------------------------
extern "C" void kernel_launch(void* const* d_in, const int* in_sizes, int n_in,
                              void* d_out, int out_size, void* d_ws, size_t ws_size,
                              hipStream_t stream)
{
    const float* inputs   = (const float*)d_in[0];   // 16,777,216 f32
    const float* codebook = (const float*)d_in[1];   //  1,048,576 f32
    float* out = (float*)d_out;                      // [0]=loss, [1..]=quantized

    // workspace carve-out (~4.6 MB)
    char* ws = (char*)d_ws;
    bf16_t* cb_hi    = (bf16_t*)(ws);                                   // 2 MB
    bf16_t* cb_lo    = (bf16_t*)(ws + (2u << 20));                      // 2 MB
    float*  c_sq     = (float*) (ws + (4u << 20));                      // 16 KB
    int*    idxbuf   = (int*)   (ws + (4u << 20) + (64u << 10));        // 256 KB
    float*  partials = (float*) (ws + (4u << 20) + (64u << 10) + (256u << 10)); // 256 KB

    // 0) codebook split + norms: 4096 code rows, 8 rows (waves) per block
    vq_prep_kernel<<<(HEADS * KCODES) / 8, 256, 0, stream>>>(codebook, cb_hi, cb_lo, c_sq);

    // 1) WMMA distance argmin: 128 rows per block, grid (16384/128, 4 heads)
    vq_argmin_kernel<<<dim3(NROWS / 128, HEADS), 256, 0, stream>>>(
        inputs, cb_hi, cb_lo, c_sq, idxbuf);

    // 2) gather + SSE partials
    vq_gather_kernel<<<GATHER_BLOCKS, 256, 0, stream>>>(
        inputs, codebook, idxbuf, out + 1, partials);

    // 3) loss
    vq_loss_kernel<<<1, 256, 0, stream>>>(partials, out);
}
